// ParticleNet_86397562127113
// MI455X (gfx1250) — matrix operand order repaired and checked by reading, not compiled, not run on hardware
//
#include <hip/hip_runtime.h>
#include <math.h>

typedef __attribute__((ext_vector_type(16))) _Float16 v16h;
typedef __attribute__((ext_vector_type(8)))  float    v8f;

#define ACT_NONE  0
#define ACT_RELU  1
#define ACT_SPLIT 2   // col < 512 -> sigmoid, else tanh (encoder head)

namespace {
constexpr int BN        = 256;            // batch
constexpr int IN_DIM    = 128;
constexpr int HD        = 128;
constexpr int NPART     = 256;
constexpr int KP        = 8;
constexpr int KNBR      = 16;
constexpr int TSTEPS    = 10;
constexpr int STATE_DIM = NPART * (2 + KP);   // 2560
constexpr int OUT_DIM   = 64;

constexpr int MT = 32;     // block tile M
constexpr int NT = 128;    // block tile N
constexpr int KT = 64;     // block tile K (2 wmma K-steps)
}

// ---------------------------------------------------------------- f32 -> f16
__global__ void __launch_bounds__(256)
cvt_f32_f16(const float* __restrict__ in, _Float16* __restrict__ out, int n) {
  int i = blockIdx.x * 256 + threadIdx.x;
  if (i < n) out[i] = (_Float16)in[i];
}

// ------------------------------------------------------ WMMA GEMM + epilogue
// C(M,N) = A(M,K) * W(K,N) + bias, optional activation.
// Block: 256 threads = 8 wave32s in a 2(M) x 4(N) grid; each wave owns a
// 16x32 output (two 16x16 WMMA accumulators). K-tile 64 = 2 wmma K-steps,
// so 4 static v_wmma sites per staged tile. B tile stored transposed in LDS
// so every lane's 16 B-halves are two contiguous ds_load_b128 runs.
__global__ void __launch_bounds__(256)
wmma_gemm(const _Float16* __restrict__ A, const _Float16* __restrict__ W,
          const float* __restrict__ bias, float* __restrict__ outF,
          _Float16* __restrict__ outH, int M, int N, int K, int act)
{
  __shared__ _Float16 As[MT][KT];        // [32][64]   row-major, 4 KB
  __shared__ _Float16 Bs[NT][KT + 8];    // [128][72]  col-major(k), 18 KB

  const int tid   = threadIdx.x;
  const int wave  = tid >> 5;
  const int lane  = tid & 31;
  const int group = lane >> 4;           // lane half: 0 or 1
  const int lhalf = lane & 15;
  const int waveM = wave >> 2;           // 0..1
  const int waveN = wave & 3;            // 0..3

  const int blockrow = blockIdx.y * MT;
  const int colblock = blockIdx.x * NT;
  const int rowbase  = blockrow + waveM * 16;
  const int c0 = colblock + waveN * 32;  // this wave's two 16-col subtiles
  const int c1 = c0 + 16;
  const bool act0 = (c0 < N);            // uniform per wave
  const bool act1 = (c1 < N);

  v8f acc0 = {}, acc1 = {};

  for (int k0 = 0; k0 < K; k0 += KT) {
    // ---- stage A tile 32x64 (2048 halves, 8/thread), coalesced in k ----
    for (int i = tid; i < MT * KT; i += 256) {
      int r = i >> 6, c = i & 63;
      As[r][c] = A[(size_t)(blockrow + r) * K + (k0 + c)];
    }
    // ---- stage B tile 64x128 transposed (8192 halves, 32/thread) ----
    for (int i = tid; i < KT * NT; i += 256) {
      int r = i >> 7, c = i & 127;       // r = k-within-tile, c = col
      int gc = colblock + c;
      Bs[c][r] = (gc < N) ? W[(size_t)(k0 + r) * N + gc] : (_Float16)0.0f;
    }
    // prefetch next K-tile toward the WGP while this one is consumed
    if (k0 + KT < K) {
      __builtin_prefetch(&A[(size_t)(blockrow + (tid >> 3)) * K + (k0 + KT)], 0, 1);
      __builtin_prefetch(&W[(size_t)(k0 + KT + (tid >> 2)) * N + colblock], 0, 1);
    }
    __syncthreads();

#pragma unroll
    for (int s = 0; s < 2; ++s) {        // two 16x16x32 wmma K-steps
      v16h af, bf0, bf1;
      // A fragment (16-bit A 16x32): lane row = lhalf;
      // halves = one contiguous 16-half run at k = group*16 + 0..15.
#pragma unroll
      for (int h = 0; h < 16; ++h) {
        int k = s * 32 + (group << 4) + h;
        af[h] = As[waveM * 16 + lhalf][k];
      }
      // B fragments: column = lhalf of subtile; lanes 0-15 K=0..15,
      // lanes 16-31 K=16..31 -> 16 contiguous halves in transposed Bs.
#pragma unroll
      for (int h = 0; h < 16; ++h)
        bf0[h] = Bs[waveN * 32 + lhalf][s * 32 + (group << 4) + h];
#pragma unroll
      for (int h = 0; h < 16; ++h)
        bf1[h] = Bs[waveN * 32 + 16 + lhalf][s * 32 + (group << 4) + h];

      if (act0)
        acc0 = __builtin_amdgcn_wmma_f32_16x16x32_f16(
            false, af, false, bf0, (short)0, acc0, false, false);
      if (act1)
        acc1 = __builtin_amdgcn_wmma_f32_16x16x32_f16(
            false, af, false, bf1, (short)0, acc1, false, false);
    }
    __syncthreads();
  }

  // ---- epilogue: bias + activation + store (f32 and/or f16) ----
#pragma unroll
  for (int sub = 0; sub < 2; ++sub) {
    const bool on = sub ? act1 : act0;
    if (!on) continue;
    const int cbase = sub ? c1 : c0;
    const v8f acc = sub ? acc1 : acc0;
#pragma unroll
    for (int e = 0; e < 8; ++e) {
      int m = rowbase + (group << 3) + e;   // C/D: VGPR e -> row group*8+e
      int n = cbase + lhalf;
      float val = acc[e] + bias[n];
      if (act == ACT_RELU) {
        val = fmaxf(val, 0.0f);
      } else if (act == ACT_SPLIT) {
        val = (n < 2 * NPART) ? (1.0f / (1.0f + expf(-val))) : tanhf(val);
      }
      if (outF) outF[(size_t)m * N + n] = val;
      if (outH) outH[(size_t)m * N + n] = (_Float16)val;
    }
  }
}

// ----------------------------------------------------------- particle sim
__device__ __forceinline__ float pwrap(float d) {
  return d - rintf(d);          // SPACE == 1.0, ties-to-even like jnp.round
}

// One block per batch element; thread i owns particle i. Positions + prefs
// live in LDS across the 10 steps; top-16 KNN held in registers.
__global__ void __launch_bounds__(256)
sim_kernel(const float* __restrict__ state, _Float16* __restrict__ dec_in)
{
  __shared__ float sx[NPART];
  __shared__ float sy[NPART];
  __shared__ float sp[NPART][KP];

  const int b = blockIdx.x;
  const int i = threadIdx.x;
  const float* srow = state + (size_t)b * STATE_DIM;

  sx[i] = srow[2 * i + 0];
  sy[i] = srow[2 * i + 1];
  float myp[KP];
#pragma unroll
  for (int k = 0; k < KP; ++k) {
    float p = srow[2 * NPART + i * KP + k];
    sp[i][k] = p;
    myp[k] = p;      // SOCIAL == 0: prefs are constant over time
  }
  __syncthreads();

  for (int t = 0; t < TSTEPS; ++t) {
    const float px = sx[i], py = sy[i];

    // --- 16-nearest-neighbor selection (sorted ascending, reg-resident) ---
    float nd[KNBR];
    int   ni[KNBR];
#pragma unroll
    for (int q = 0; q < KNBR; ++q) { nd[q] = 1e30f; ni[q] = 0; }

    for (int j = 0; j < NPART; ++j) {
      if (j == i) continue;
      float dx = pwrap(sx[j] - px);
      float dy = pwrap(sy[j] - py);
      float dist = sqrtf(dx * dx + dy * dy);
      if (dist < nd[KNBR - 1]) {
        nd[KNBR - 1] = dist;
        ni[KNBR - 1] = j;
#pragma unroll
        for (int q = KNBR - 1; q > 0; --q) {   // bubble into place
          if (nd[q] < nd[q - 1]) {
            float td = nd[q]; nd[q] = nd[q - 1]; nd[q - 1] = td;
            int   ti = ni[q]; ni[q] = ni[q - 1]; ni[q - 1] = ti;
          }
        }
      }
    }

    // --- neighbor interaction ---
    float movx = 0.f, movy = 0.f, pushx = 0.f, pushy = 0.f;
#pragma unroll 4
    for (int q = 0; q < KNBR; ++q) {
      int j = ni[q];
      float dx = pwrap(sx[j] - px);
      float dy = pwrap(sy[j] - py);
      float nrm = sqrtf(dx * dx + dy * dy);
      float d   = fmaxf(nrm, 1e-12f);
      float ux  = dx / d, uy = dy / d;
      float ip = 0.f;
#pragma unroll
      for (int k = 0; k < KP; ++k) ip += myp[k] * sp[j][k];
      ip *= (1.0f / KP);
      movx += ip * ux;  movy += ip * uy;
      float dd = fmaxf(d, 1e-6f);
      pushx -= ux / dd; pushy -= uy / dd;
    }
    const float inv = 1.0f / KNBR;
    float nx = px + 0.01f * (movx * inv) + 0.001f * (pushx * inv);
    float ny = py + 0.01f * (movy * inv) + 0.001f * (pushy * inv);
    nx -= floorf(nx);       // mod 1.0 into [0,1)
    ny -= floorf(ny);

    __syncthreads();        // all reads of step t done
    sx[i] = nx; sy[i] = ny;
    __syncthreads();
  }

  // emit decoder input in f16: [pos.flat(512) | prefs.flat(2048)]
  _Float16* drow = dec_in + (size_t)b * STATE_DIM;
  drow[2 * i + 0] = (_Float16)sx[i];
  drow[2 * i + 1] = (_Float16)sy[i];
#pragma unroll
  for (int k = 0; k < KP; ++k)
    drow[2 * NPART + i * KP + k] = (_Float16)myp[k];
}

// ---------------------------------------------------------------- launcher
extern "C" void kernel_launch(void* const* d_in, const int* in_sizes, int n_in,
                              void* d_out, int out_size, void* d_ws, size_t ws_size,
                              hipStream_t stream) {
  (void)in_sizes; (void)n_in; (void)out_size; (void)ws_size;

  const float* x      = (const float*)d_in[0];
  const float* enc_w1 = (const float*)d_in[1];
  const float* enc_b1 = (const float*)d_in[2];
  const float* enc_w2 = (const float*)d_in[3];
  const float* enc_b2 = (const float*)d_in[4];
  const float* enc_w3 = (const float*)d_in[5];
  const float* enc_b3 = (const float*)d_in[6];
  const float* dec_w1 = (const float*)d_in[7];
  const float* dec_b1 = (const float*)d_in[8];
  const float* dec_w2 = (const float*)d_in[9];
  const float* dec_b2 = (const float*)d_in[10];
  const float* dec_w3 = (const float*)d_in[11];
  const float* dec_b3 = (const float*)d_in[12];

  char* base = (char*)d_ws;
  size_t off = 0;
  auto alloc = [&](size_t elems, size_t esz) -> void* {
    void* p = base + off;
    off += (elems * esz + 255) & ~(size_t)255;
    return p;
  };

  _Float16* xh    = (_Float16*)alloc((size_t)BN * IN_DIM, 2);
  _Float16* we1h  = (_Float16*)alloc((size_t)IN_DIM * HD, 2);
  _Float16* we2h  = (_Float16*)alloc((size_t)HD * HD, 2);
  _Float16* we3h  = (_Float16*)alloc((size_t)HD * STATE_DIM, 2);
  _Float16* wd1h  = (_Float16*)alloc((size_t)STATE_DIM * HD, 2);
  _Float16* wd2h  = (_Float16*)alloc((size_t)HD * HD, 2);
  _Float16* wd3h  = (_Float16*)alloc((size_t)HD * OUT_DIM, 2);
  _Float16* h1h   = (_Float16*)alloc((size_t)BN * HD, 2);
  _Float16* h2h   = (_Float16*)alloc((size_t)BN * HD, 2);
  float*    state = (float*)   alloc((size_t)BN * STATE_DIM, 4);
  _Float16* dinh  = (_Float16*)alloc((size_t)BN * STATE_DIM, 2);
  _Float16* d1h   = (_Float16*)alloc((size_t)BN * HD, 2);
  _Float16* d2h   = (_Float16*)alloc((size_t)BN * HD, 2);

  auto cdiv = [](int a, int b) { return (a + b - 1) / b; };

  // f32 -> f16 staging (x + all six weight matrices)
  cvt_f32_f16<<<cdiv(BN * IN_DIM, 256),        256, 0, stream>>>(x,      xh,   BN * IN_DIM);
  cvt_f32_f16<<<cdiv(IN_DIM * HD, 256),        256, 0, stream>>>(enc_w1, we1h, IN_DIM * HD);
  cvt_f32_f16<<<cdiv(HD * HD, 256),            256, 0, stream>>>(enc_w2, we2h, HD * HD);
  cvt_f32_f16<<<cdiv(HD * STATE_DIM, 256),     256, 0, stream>>>(enc_w3, we3h, HD * STATE_DIM);
  cvt_f32_f16<<<cdiv(STATE_DIM * HD, 256),     256, 0, stream>>>(dec_w1, wd1h, STATE_DIM * HD);
  cvt_f32_f16<<<cdiv(HD * HD, 256),            256, 0, stream>>>(dec_w2, wd2h, HD * HD);
  cvt_f32_f16<<<cdiv(HD * OUT_DIM, 256),       256, 0, stream>>>(dec_w3, wd3h, HD * OUT_DIM);

  // encoder
  wmma_gemm<<<dim3(cdiv(HD, NT),        BN / MT), 256, 0, stream>>>(
      xh,  we1h, enc_b1, nullptr, h1h, BN, HD,        IN_DIM, ACT_RELU);
  wmma_gemm<<<dim3(cdiv(HD, NT),        BN / MT), 256, 0, stream>>>(
      h1h, we2h, enc_b2, nullptr, h2h, BN, HD,        HD,     ACT_RELU);
  wmma_gemm<<<dim3(cdiv(STATE_DIM, NT), BN / MT), 256, 0, stream>>>(
      h2h, we3h, enc_b3, state, nullptr, BN, STATE_DIM, HD,   ACT_SPLIT);

  // particle simulation (10 steps, LDS-resident)
  sim_kernel<<<BN, NPART, 0, stream>>>(state, dinh);

  // decoder
  wmma_gemm<<<dim3(cdiv(HD, NT),      BN / MT), 256, 0, stream>>>(
      dinh, wd1h, dec_b1, nullptr, d1h, BN, HD,      STATE_DIM, ACT_RELU);
  wmma_gemm<<<dim3(cdiv(HD, NT),      BN / MT), 256, 0, stream>>>(
      d1h,  wd2h, dec_b2, nullptr, d2h, BN, HD,      HD,        ACT_RELU);
  wmma_gemm<<<dim3(cdiv(OUT_DIM, NT), BN / MT), 256, 0, stream>>>(
      d2h,  wd3h, dec_b3, (float*)d_out, nullptr, BN, OUT_DIM, HD, ACT_NONE);
}
